// SwinDecoder_67422396613086
// MI455X (gfx1250) — matrix operand order repaired and checked by reading, not compiled, not run on hardware
//
#include <hip/hip_runtime.h>

typedef __attribute__((ext_vector_type(16))) __bf16 bf16x16;
typedef __attribute__((ext_vector_type(8)))  float  v8f;

#define DEV __device__ __forceinline__

static constexpr float ATTN_SCALE = 0.17677669529663687f; // 1/sqrt(32)

DEV v8f zero8() { v8f z = {0.f,0.f,0.f,0.f,0.f,0.f,0.f,0.f}; return z; }

DEV v8f wmma_bf16(bf16x16 a, bf16x16 b, v8f c) {
  return __builtin_amdgcn_wmma_f32_16x16x32_bf16(false, a, false, b, (short)0, c, false, false);
}

// async global -> LDS 16B copy (CDNA5 GLOBAL_LOAD_ASYNC_TO_LDS_B128, ASYNCcnt-tracked)
DEV void async_ld128(void* lds_base, void* lds_dst, const void* gsrc) {
  unsigned lds_off = (unsigned)((const char*)lds_dst - (const char*)lds_base);
  unsigned long long ga = (unsigned long long)(uintptr_t)gsrc;
  asm volatile("global_load_async_to_lds_b128 %0, %1, off"
               :: "v"(lds_off), "v"(ga) : "memory");
}
DEV void wait_async0() { asm volatile("s_wait_asynccnt 0" ::: "memory"); }

// A-fragment (16x32 bf16), S row-major [row][chan], stride ld (bf16 elems).
// Lane L: m = m0 + (L&15); elems 0..7 -> K = kb+0..7, elems 8..15 -> K = kb+16..23, kb = (L>>4)*8.
DEV bf16x16 ldsA(const __bf16* S, int ld, int m0, int k0) {
  int l = threadIdx.x & 31;
  const __bf16* p = S + (m0 + (l & 15)) * ld + k0 + ((l >> 4) << 3);
  bf16x16 a;
#pragma unroll
  for (int e = 0; e < 8; ++e) a[e] = p[e];
#pragma unroll
  for (int e = 0; e < 8; ++e) a[8 + e] = p[16 + e];
  return a;
}

// B-fragment (32x16) with B[k][n] = S[n][k]  (n indexes rows of S): two 8-elem reads.
// Lane L: n = n0 + (L&15); elems 0..7 -> K = kb+0..7, elems 8..15 -> K = kb+16..23, kb = (L>>4)*8.
DEV bf16x16 ldsBT(const __bf16* S, int ld, int n0, int k0) {
  int l = threadIdx.x & 31;
  const __bf16* p = S + (n0 + (l & 15)) * ld + k0 + ((l >> 4) << 3);
  bf16x16 b;
#pragma unroll
  for (int e = 0; e < 8; ++e) b[e] = p[e];
#pragma unroll
  for (int e = 0; e < 8; ++e) b[8 + e] = p[16 + e];
  return b;
}

// B-fragment from pre-converted bf16 weight W[o][c] (row-major, ld elems): B[c][o] = W[o][c].
DEV bf16x16 gWb(const __bf16* W, int ld, int n0, int k0) {
  int l = threadIdx.x & 31;
  const __bf16* p = W + (size_t)(n0 + (l & 15)) * ld + k0 + ((l >> 4) << 3);
  uint4 lo = *(const uint4*)p, hi = *(const uint4*)(p + 16);
  union { uint4 u[2]; bf16x16 v; } r;
  r.u[0] = lo; r.u[1] = hi;
  return r.v;
}

// ---------------------------------------------------------------------------
// Prep kernel: one-shot f32->bf16 weight conversion + rel-pos bias table
// biasT[h][k][q] = rpb[rel_idx(q,k)*4 + h]   (window-independent)
// ---------------------------------------------------------------------------
__global__ __launch_bounds__(256) void k_prep(
    const float* __restrict__ qw, const float* __restrict__ kvw,
    const float* __restrict__ pw, const float* __restrict__ f1w,
    const float* __restrict__ f2w, const float* __restrict__ rpb,
    __bf16* __restrict__ bqw, __bf16* __restrict__ bkvw, __bf16* __restrict__ bpw,
    __bf16* __restrict__ bf1, __bf16* __restrict__ bf2, float* __restrict__ biasT)
{
  int i = blockIdx.x * 256 + threadIdx.x;
  if (i < 16384) bqw[i] = (__bf16)qw[i];
  if (i < 32768) bkvw[i] = (__bf16)kvw[i];
  if (i < 16384) bpw[i] = (__bf16)pw[i];
  if (i < 65536) { bf1[i] = (__bf16)f1w[i]; bf2[i] = (__bf16)f2w[i]; }
  if (i < 16384) {
    int q = i & 63, k = (i >> 6) & 63, h = i >> 12;
    int qi = q >> 3, qj = q & 7, ki = k >> 3, kj = k & 7;
    int ridx = (qi - ki + 7) * 15 + (qj - kj + 7);
    biasT[i] = rpb[ridx * 4 + h];
  }
}

// ---------------------------------------------------------------------------
// Kernel 0: channels-last transpose + LayerNorm1.  WG = 64 pixels of one row.
// ---------------------------------------------------------------------------
__global__ __launch_bounds__(128) void k0_ln(
    const float* __restrict__ x, const float* __restrict__ kv,
    const float* __restrict__ g, const float* __restrict__ be,
    float* __restrict__ xp, __bf16* __restrict__ xn, __bf16* __restrict__ kvn)
{
  extern __shared__ char smem[];
  float* sT    = (float*)smem;        // [2][128*64]
  float* sMean = sT + 2 * 8192;       // [2][64]
  float* sRstd = sMean + 128;         // [2][64]

  int blk = blockIdx.x;
  int b = blk >> 10; int rem = blk & 1023;
  int hr = rem >> 2; int w0 = (rem & 3) << 6;
  int tid = threadIdx.x;

  // stage both 128ch x 64px tiles via async global->LDS DMA
  for (int s = 0; s < 2; ++s) {
    const float* src = s ? kv : x;
    for (int i = tid; i < 2048; i += 128) {
      int c = i >> 4; int p4 = (i & 15) << 2;
      const float* gp = src + (size_t)(b * 128 + c) * 65536 + hr * 256 + w0 + p4;
      async_ld128(smem, sT + s * 8192 + c * 64 + p4, gp);
    }
  }
  wait_async0();
  __syncthreads();

  { // per-pixel stats: threads 0..63 -> tensor0, 64..127 -> tensor1
    int s = tid >> 6; int p = tid & 63;
    const float* T = sT + s * 8192;
    float sum = 0.f, sq = 0.f;
#pragma unroll 8
    for (int c = 0; c < 128; ++c) { float v = T[c * 64 + p]; sum += v; sq += v * v; }
    float m = sum * 0.0078125f;
    float var = sq * 0.0078125f - m * m;
    sMean[s * 64 + p] = m;
    sRstd[s * 64 + p] = rsqrtf(var + 1e-5f);
  }
  __syncthreads();

  int p = tid >> 1; int c0 = (tid & 1) << 6;
  size_t pix = ((size_t)(b * 256 + hr)) * 256 + w0 + p;
  for (int s = 0; s < 2; ++s) {
    const float* T = sT + s * 8192;
    float m = sMean[s * 64 + p], rs = sRstd[s * 64 + p];
    __bf16* dstn = s ? kvn : xn;
#pragma unroll 8
    for (int i = 0; i < 64; ++i) {
      int c = c0 + i;
      float v = T[c * 64 + p];
      dstn[pix * 128 + c] = (__bf16)((v - m) * rs * g[c] + be[c]);
      if (s == 0) xp[pix * 128 + c] = v;
    }
  }
}

// ---------------------------------------------------------------------------
// Kernel 1: fused windowed attention.  WG = one 8x8 window; wave w = head w.
// ---------------------------------------------------------------------------
#define LDA 136   // padded row stride (bf16) for 64x128 token-major tiles
#define LDV 72    // padded row stride (bf16) for 128x64 channel-major V^T tile

__global__ __launch_bounds__(128) void k1_attn(
    const __bf16* __restrict__ xn, const __bf16* __restrict__ kvn,
    const float* __restrict__ xp,
    const __bf16* __restrict__ qw, const float* __restrict__ qb,
    const __bf16* __restrict__ kvw, const float* __restrict__ kvb,
    const float* __restrict__ biasT,
    const __bf16* __restrict__ pw, const float* __restrict__ pb,
    float* __restrict__ xo)
{
  extern __shared__ char smem[];
  __bf16* sX  = (__bf16*)smem;        // 64 x LDA (token-major)
  __bf16* sKV = sX  + 64 * LDA;       // 64 x LDA
  __bf16* sQ  = sKV + 64 * LDA;       // 64 x LDA
  __bf16* sK  = sQ  + 64 * LDA;       // 64 x LDA
  __bf16* sVT = sK  + 64 * LDA;       // 128 x LDV (channel-major V^T)
  __bf16* sO  = sVT + 128 * LDV;      // 64 x LDA

  int blk = blockIdx.x;
  int b = blk >> 10; int wid = blk & 1023;
  int wy = wid >> 5, wx = wid & 31;
  int tid = threadIdx.x, lane = tid & 31, h = tid >> 5;
  int halfr = (lane >> 4) << 3;

  // ---- stage Xn / KVn windows via async global->LDS DMA (roll folded into gather) ----
  for (int i = tid; i < 1024; i += 128) {
    int t = i & 511;
    int tok = t >> 3, ch = t & 7;
    int qi = tok >> 3, qj = tok & 7;
    int gh = (wy * 8 + qi + 4) & 255;
    int gw = (wx * 8 + qj + 4) & 255;
    size_t pix = ((size_t)(b * 256 + gh)) * 256 + gw;
    const __bf16* src = ((i < 512) ? xn : kvn) + pix * 128 + ch * 8;
    __bf16* dst = ((i < 512) ? sX : sKV) + tok * LDA + ch * 8;
    async_ld128(smem, dst, src);
  }
  wait_async0();
  __syncthreads();

  // ---- Q projection for this wave's head (channels h*32 .. h*32+31) ----
  {
    v8f accQ[4][2];
#pragma unroll
    for (int mt = 0; mt < 4; ++mt) { accQ[mt][0] = zero8(); accQ[mt][1] = zero8(); }
#pragma unroll
    for (int kt = 0; kt < 4; ++kt) {
      bf16x16 aX[4];
#pragma unroll
      for (int mt = 0; mt < 4; ++mt) aX[mt] = ldsA(sX, LDA, mt * 16, kt * 32);
#pragma unroll
      for (int j = 0; j < 2; ++j) {
        bf16x16 bq = gWb(qw, 128, h * 32 + j * 16, kt * 32);
#pragma unroll
        for (int mt = 0; mt < 4; ++mt) accQ[mt][j] = wmma_bf16(aX[mt], bq, accQ[mt][j]);
      }
    }
#pragma unroll
    for (int mt = 0; mt < 4; ++mt)
#pragma unroll
      for (int j = 0; j < 2; ++j) {
        int o = h * 32 + j * 16 + (lane & 15);
        int tb = mt * 16 + halfr;
#pragma unroll
        for (int r = 0; r < 8; ++r)
          sQ[(tb + r) * LDA + o] = (__bf16)(accQ[mt][j][r] + qb[o]);
      }
  }
  // ---- K/V projections (V stored transposed, packed b128 stores) ----
  {
    v8f accK[4][2], accV[4][2];
#pragma unroll
    for (int mt = 0; mt < 4; ++mt)
#pragma unroll
      for (int j = 0; j < 2; ++j) { accK[mt][j] = zero8(); accV[mt][j] = zero8(); }
#pragma unroll
    for (int kt = 0; kt < 4; ++kt) {
      bf16x16 aKV[4];
#pragma unroll
      for (int mt = 0; mt < 4; ++mt) aKV[mt] = ldsA(sKV, LDA, mt * 16, kt * 32);
#pragma unroll
      for (int j = 0; j < 2; ++j) {
        int n0 = h * 32 + j * 16;
        bf16x16 bk = gWb(kvw, 128, n0,       kt * 32);
        bf16x16 bv = gWb(kvw, 128, 128 + n0, kt * 32);
#pragma unroll
        for (int mt = 0; mt < 4; ++mt) {
          accK[mt][j] = wmma_bf16(aKV[mt], bk, accK[mt][j]);
          accV[mt][j] = wmma_bf16(aKV[mt], bv, accV[mt][j]);
        }
      }
    }
#pragma unroll
    for (int mt = 0; mt < 4; ++mt)
#pragma unroll
      for (int j = 0; j < 2; ++j) {
        int o = h * 32 + j * 16 + (lane & 15);
        int tb = mt * 16 + halfr;
#pragma unroll
        for (int r = 0; r < 8; ++r)
          sK[(tb + r) * LDA + o] = (__bf16)(accK[mt][j][r] + kvb[o]);
        union { uint4 u; __bf16 e[8]; } pk;
#pragma unroll
        for (int r = 0; r < 8; ++r)
          pk.e[r] = (__bf16)(accV[mt][j][r] + kvb[128 + o]);
        *(uint4*)(sVT + o * LDV + tb) = pk.u;   // V^T row = channel, 8 consecutive tokens
      }
  }

  // ---- S^T = K * Q^T  (m = k-token, n = q-token) ----
  v8f st[4][4];
  {
    bf16x16 aK[4], bQ[4];
#pragma unroll
    for (int t = 0; t < 4; ++t) {
      aK[t] = ldsA (sK, LDA, t * 16, h * 32);
      bQ[t] = ldsBT(sQ, LDA, t * 16, h * 32);
    }
#pragma unroll
    for (int kt = 0; kt < 4; ++kt)
#pragma unroll
      for (int qt = 0; qt < 4; ++qt) st[kt][qt] = wmma_bf16(aK[kt], bQ[qt], zero8());
  }

  // epilogue: scale + precomputed rel-pos bias (biasT[h][k][q]); coalesced over q
#pragma unroll
  for (int qt = 0; qt < 4; ++qt) {
    int q = qt * 16 + (lane & 15);
#pragma unroll
    for (int kt = 0; kt < 4; ++kt)
#pragma unroll
      for (int r = 0; r < 8; ++r) {
        int k = kt * 16 + halfr + r;
        st[kt][qt][r] = st[kt][qt][r] * ATTN_SCALE + biasT[(h * 64 + k) * 64 + q];
      }
  }
  // shift mask only for edge windows (wave-uniform branch; 63/1024 windows)
  if (wy == 31 || wx == 31) {
#pragma unroll
    for (int qt = 0; qt < 4; ++qt) {
      int q = qt * 16 + (lane & 15);
      int ghq = wy * 8 + (q >> 3), gwq = wx * 8 + (q & 7);
      int rq = (ghq < 248 ? 0 : (ghq < 252 ? 1 : 2)) * 3 + (gwq < 248 ? 0 : (gwq < 252 ? 1 : 2));
#pragma unroll
      for (int kt = 0; kt < 4; ++kt)
#pragma unroll
        for (int r = 0; r < 8; ++r) {
          int k = kt * 16 + halfr + r;
          int ghk = wy * 8 + (k >> 3), gwk = wx * 8 + (k & 7);
          int rk = (ghk < 248 ? 0 : (ghk < 252 ? 1 : 2)) * 3 + (gwk < 248 ? 0 : (gwk < 252 ? 1 : 2));
          if (rq != rk) st[kt][qt][r] -= 100.f;
        }
    }
  }

  // column softmax over k: in-lane over 32 regs + one xor-16 shuffle
  float inv[4];
#pragma unroll
  for (int qt = 0; qt < 4; ++qt) {
    float m = -1e30f;
#pragma unroll
    for (int kt = 0; kt < 4; ++kt)
#pragma unroll
      for (int r = 0; r < 8; ++r) m = fmaxf(m, st[kt][qt][r]);
    m = fmaxf(m, __shfl_xor(m, 16));
    float s = 0.f;
#pragma unroll
    for (int kt = 0; kt < 4; ++kt)
#pragma unroll
      for (int r = 0; r < 8; ++r) { float e = __expf(st[kt][qt][r] - m); st[kt][qt][r] = e; s += e; }
    s += __shfl_xor(s, 16);
    inv[qt] = 1.f / s;
  }

  // S^T C-layout re-packs directly into P A-fragments (no LDS transpose)
  bf16x16 aP[4][2];
#pragma unroll
  for (int qt = 0; qt < 4; ++qt)
#pragma unroll
    for (int k2 = 0; k2 < 2; ++k2) {
#pragma unroll
      for (int e = 0; e < 8; ++e) aP[qt][k2][e]     = (__bf16)(st[2 * k2][qt][e]     * inv[qt]);
#pragma unroll
      for (int e = 0; e < 8; ++e) aP[qt][k2][8 + e] = (__bf16)(st[2 * k2 + 1][qt][e] * inv[qt]);
    }

  // O = P * V  (V^T rows are channels -> contiguous B-fragments)
  v8f o[4][2];
#pragma unroll
  for (int qt = 0; qt < 4; ++qt) { o[qt][0] = zero8(); o[qt][1] = zero8(); }
#pragma unroll
  for (int k2 = 0; k2 < 2; ++k2) {
    bf16x16 bV0 = ldsBT(sVT, LDV, h * 32,      k2 * 32);
    bf16x16 bV1 = ldsBT(sVT, LDV, h * 32 + 16, k2 * 32);
#pragma unroll
    for (int qt = 0; qt < 4; ++qt) {
      o[qt][0] = wmma_bf16(aP[qt][k2], bV0, o[qt][0]);
      o[qt][1] = wmma_bf16(aP[qt][k2], bV1, o[qt][1]);
    }
  }
#pragma unroll
  for (int qt = 0; qt < 4; ++qt)
#pragma unroll
    for (int nt = 0; nt < 2; ++nt)
#pragma unroll
      for (int r = 0; r < 8; ++r)
        sO[(qt * 16 + halfr + r) * LDA + h * 32 + nt * 16 + (lane & 15)] = (__bf16)o[qt][nt][r];

  __syncthreads();

  // ---- proj GEMM + bias + shortcut -> xo (BHWC f32) ----
  v8f accP[4][2];
#pragma unroll
  for (int mt = 0; mt < 4; ++mt) { accP[mt][0] = zero8(); accP[mt][1] = zero8(); }
#pragma unroll
  for (int kt = 0; kt < 4; ++kt) {
    bf16x16 aO[4];
#pragma unroll
    for (int mt = 0; mt < 4; ++mt) aO[mt] = ldsA(sO, LDA, mt * 16, kt * 32);
#pragma unroll
    for (int j = 0; j < 2; ++j) {
      bf16x16 bw = gWb(pw, 128, (h * 2 + j) * 16, kt * 32);
#pragma unroll
      for (int mt = 0; mt < 4; ++mt) accP[mt][j] = wmma_bf16(aO[mt], bw, accP[mt][j]);
    }
  }
#pragma unroll
  for (int mt = 0; mt < 4; ++mt)
#pragma unroll
    for (int j = 0; j < 2; ++j) {
      int ocol = (h * 2 + j) * 16 + (lane & 15);
      int tb = mt * 16 + halfr;
#pragma unroll
      for (int r = 0; r < 8; ++r) {
        int tok = tb + r;
        int qi = tok >> 3, qj = tok & 7;
        int gh = (wy * 8 + qi + 4) & 255;
        int gw = (wx * 8 + qj + 4) & 255;
        size_t pix = ((size_t)(b * 256 + gh)) * 256 + gw;
        xo[pix * 128 + ocol] = accP[mt][j][r] + pb[ocol] + xp[pix * 128 + ocol];
      }
    }
}

// ---------------------------------------------------------------------------
// Kernel 2: LN2 + MLP (fc1 -> exact GELU -> fc2) + residual -> out (B,C,H,W)
// ---------------------------------------------------------------------------
#define LDH 520   // padded row stride (bf16) for 64x512 hidden tile

__global__ __launch_bounds__(128) void k2_mlp(
    const float* __restrict__ xo,
    const float* __restrict__ g2, const float* __restrict__ b2,
    const __bf16* __restrict__ f1w, const float* __restrict__ f1b,
    const __bf16* __restrict__ f2w, const float* __restrict__ f2b,
    float* __restrict__ out)
{
  extern __shared__ char smem[];
  __bf16* sX = (__bf16*)smem;      // 64 x LDA
  __bf16* sH = sX + 64 * LDA;      // 64 x LDH

  int blk = blockIdx.x;
  int b = blk >> 10; int rem = blk & 1023;
  int hr = rem >> 2; int w0 = (rem & 3) << 6;
  int tid = threadIdx.x, lane = tid & 31, wv = tid >> 5;
  int halfr = (lane >> 4) << 3;

  { // LayerNorm2 -> bf16 sX ; thread t handles pixel t/2, channel half (t&1)
    int p = tid >> 1, c0 = (tid & 1) << 6;
    size_t pix = ((size_t)(b * 256 + hr)) * 256 + w0 + p;
    const float* src = xo + pix * 128 + c0;
    float vbuf[64];
    float sum = 0.f, sq = 0.f;
#pragma unroll
    for (int i = 0; i < 64; ++i) { float v = src[i]; vbuf[i] = v; sum += v; sq += v * v; }
    sum += __shfl_xor(sum, 1);
    sq  += __shfl_xor(sq, 1);
    float m = sum * 0.0078125f;
    float rs = rsqrtf(sq * 0.0078125f - m * m + 1e-5f);
#pragma unroll
    for (int i = 0; i < 64; ++i) {
      int c = c0 + i;
      sX[p * LDA + c] = (__bf16)((vbuf[i] - m) * rs * g2[c] + b2[c]);
    }
  }
  __syncthreads();

  // GEMM1: (64x128)@(128x512) + GELU -> sH ; two 4-tile halves to stay <256 VGPRs
  for (int half = 0; half < 2; ++half) {
    v8f acc[4][4];
#pragma unroll
    for (int mt = 0; mt < 4; ++mt)
#pragma unroll
      for (int j = 0; j < 4; ++j) acc[mt][j] = zero8();
#pragma unroll
    for (int kt = 0; kt < 4; ++kt) {
      bf16x16 aX[4];
#pragma unroll
      for (int mt = 0; mt < 4; ++mt) aX[mt] = ldsA(sX, LDA, mt * 16, kt * 32);
#pragma unroll
      for (int j = 0; j < 4; ++j) {
        bf16x16 bw = gWb(f1w, 128, (wv * 8 + half * 4 + j) * 16, kt * 32);
#pragma unroll
        for (int mt = 0; mt < 4; ++mt) acc[mt][j] = wmma_bf16(aX[mt], bw, acc[mt][j]);
      }
    }
#pragma unroll
    for (int mt = 0; mt < 4; ++mt)
#pragma unroll
      for (int j = 0; j < 4; ++j) {
        int ocol = (wv * 8 + half * 4 + j) * 16 + (lane & 15);
        int tb = mt * 16 + halfr;
#pragma unroll
        for (int r = 0; r < 8; ++r) {
          float v = acc[mt][j][r] + f1b[ocol];
          v = 0.5f * v * (1.f + erff(v * 0.70710678118654752f));   // exact GELU
          sH[(tb + r) * LDH + ocol] = (__bf16)v;
        }
      }
  }
  __syncthreads();

  { // GEMM2: (64x512)@(512x128) + bias + residual -> out (B,C,H,W)
    v8f acc[4][2];
#pragma unroll
    for (int mt = 0; mt < 4; ++mt) { acc[mt][0] = zero8(); acc[mt][1] = zero8(); }
#pragma unroll
    for (int kt = 0; kt < 16; ++kt) {
      bf16x16 aH[4];
#pragma unroll
      for (int mt = 0; mt < 4; ++mt) aH[mt] = ldsA(sH, LDH, mt * 16, kt * 32);
#pragma unroll
      for (int j = 0; j < 2; ++j) {
        bf16x16 bw = gWb(f2w, 512, (wv * 2 + j) * 16, kt * 32);
#pragma unroll
        for (int mt = 0; mt < 4; ++mt) acc[mt][j] = wmma_bf16(aH[mt], bw, acc[mt][j]);
      }
    }
#pragma unroll
    for (int mt = 0; mt < 4; ++mt)
#pragma unroll
      for (int j = 0; j < 2; ++j) {
        int ocol = (wv * 2 + j) * 16 + (lane & 15);
        int tb = mt * 16 + halfr;
#pragma unroll
        for (int r = 0; r < 8; ++r) {
          int p = tb + r;
          size_t pix = ((size_t)(b * 256 + hr)) * 256 + w0 + p;
          float v = acc[mt][j][r] + f2b[ocol] + xo[pix * 128 + ocol];
          out[(((size_t)(b * 128 + ocol)) * 256 + hr) * 256 + (w0 + p)] = v;
        }
      }
  }
}

// ---------------------------------------------------------------------------
extern "C" void kernel_launch(void* const* d_in, const int* in_sizes, int n_in,
                              void* d_out, int out_size, void* d_ws, size_t ws_size,
                              hipStream_t stream) {
  (void)in_sizes; (void)n_in; (void)out_size; (void)ws_size;
  const float* x   = (const float*)d_in[0];
  const float* kv  = (const float*)d_in[1];
  const float* n1g = (const float*)d_in[2];
  const float* n1b = (const float*)d_in[3];
  const float* qw  = (const float*)d_in[4];
  const float* qb  = (const float*)d_in[5];
  const float* kvw = (const float*)d_in[6];
  const float* kvb = (const float*)d_in[7];
  const float* rpb = (const float*)d_in[8];
  const float* pw  = (const float*)d_in[9];
  const float* pb  = (const float*)d_in[10];
  const float* n2g = (const float*)d_in[11];
  const float* n2b = (const float*)d_in[12];
  const float* f1w = (const float*)d_in[13];
  const float* f1b = (const float*)d_in[14];
  const float* f2w = (const float*)d_in[15];
  const float* f2b = (const float*)d_in[16];
  float* out = (float*)d_out;

  char* ws = (char*)d_ws;
  float*  xp   = (float*)ws;                                  // 64 MiB  raw shortcut, BHWC
  __bf16* xn   = (__bf16*)(ws + (size_t)67108864);            // 32 MiB  LN(x), BHWC bf16
  __bf16* kvn  = (__bf16*)(ws + (size_t)100663296);           // 32 MiB  LN(kv), BHWC bf16
  float*  xo   = (float*)(ws + (size_t)134217728);            // 64 MiB  attn residual, BHWC
  __bf16* bqw  = (__bf16*)(ws + (size_t)201326592);           // 32 KiB
  __bf16* bkvw = (__bf16*)(ws + (size_t)201359360);           // 64 KiB
  __bf16* bpw  = (__bf16*)(ws + (size_t)201424896);           // 32 KiB
  __bf16* bf1  = (__bf16*)(ws + (size_t)201457664);           // 128 KiB
  __bf16* bf2  = (__bf16*)(ws + (size_t)201588736);           // 128 KiB
  float*  biasT= (float*)(ws + (size_t)201719808);            // 64 KiB

  dim3 blk(128);
  size_t lds1 = (size_t)(5 * 64 * LDA + 128 * LDV) * 2;          // ~103 KiB
  size_t lds2 = (size_t)(64 * LDA + 64 * LDH) * 2;               // ~82 KiB
  k_prep <<<256, dim3(256), 0, stream>>>(qw, kvw, pw, f1w, f2w, rpb,
                                         bqw, bkvw, bpw, bf1, bf2, biasT);
  k0_ln  <<<2048, blk, 2 * 8192 * 4 + 256 * 4, stream>>>(x, kv, n1g, n1b, xp, xn, kvn);
  k1_attn<<<2048, blk, lds1, stream>>>(xn, kvn, xp, bqw, qb, bkvw, kvb,
                                       biasT, bpw, pb, xo);
  k2_mlp <<<2048, blk, lds2, stream>>>(xo, n2g, n2b, bf1, f1b,
                                       bf2, f2b, out);
}
